// MultiheadAttention_57131654971968
// MI455X (gfx1250) — compile-verified, matrix operand order checked by reading, not run on hardware
//
#include <hip/hip_runtime.h>
#include <stdint.h>

// ---------------- problem constants ----------------
#define EMB   1024
#define NH    16
#define HD    64
#define BATCH 2
#define QLEN  2048
#define KVLEN 2048
#define NTOK  (BATCH * QLEN)        // 4096 tokens

// ---------------- types ----------------
typedef __bf16 bf16_t;
typedef __attribute__((ext_vector_type(16))) __bf16 v16bf;
typedef __attribute__((ext_vector_type(8)))  __bf16 v8bf;
typedef __attribute__((ext_vector_type(4)))  __bf16 v4bf;
typedef __attribute__((ext_vector_type(8)))  float  v8f;
typedef __attribute__((ext_vector_type(4)))  float  v4f;

union U16 { v16bf v; bf16_t e[16]; v8bf h[2]; };

// native converts -> v_cvt_pk_bf16_f32 / v_cvt_f32_bf16 on gfx1250
__device__ inline bf16_t f2bf(float f)  { return (bf16_t)f; }
__device__ inline float  bf2f(bf16_t b) { return (float)b; }

__device__ inline v8f wmma_bf16(v16bf a, v16bf b, v8f c) {
    return __builtin_amdgcn_wmma_f32_16x16x32_bf16(
        false, a, false, b, (short)0, c, false, false);
}

// ---------------- WMMA fragment loaders ----------------
// 16-bit A/B operand layout (CDNA5 16x32 bf16): lane&15 = row (M or N),
// lane<16 holds K = {0..7, 16..23}, lane>=16 holds K = {8..15, 24..31}.
// Caller pre-applies the per-lane (row*ld + koff) offset; loader grabs
// elements p[0..7] -> frag[0..7] and p[16..23] -> frag[8..15].
// Works for both global and LDS pointers (ds_load_b128 for __shared__).
__device__ inline v16bf load16(const bf16_t* p) {
    U16 u;
    u.h[0] = *(const v8bf*)(p);
    u.h[1] = *(const v8bf*)(p + 16);
    return u.v;
}
__device__ inline v16bf load16(const float* p) {
    v8f a = *(const v8f*)(p);        // 32B (two b128 loads)
    v8f b = *(const v8f*)(p + 16);
    U16 u;
    u.h[0] = __builtin_convertvector(a, v8bf);
    u.h[1] = __builtin_convertvector(b, v8bf);
    return u.v;
}

// ---------------- wave-level NT GEMM core ----------------
// C[16, 64] += A[16, K] * B[64, K]^T   (A pre-offset to (m0,0), B to (n0,0))
template <typename TA, typename TB>
__device__ inline void wave_gemm(const TA* __restrict__ A, int lda,
                                 const TB* __restrict__ Bm, int ldb,
                                 int K, v8f* acc) {
    const int lane = threadIdx.x & 31;
    const int r    = lane & 15;
    const int koff = (lane >> 4) << 3;
    const TA* a = A + (int64_t)r * lda + koff;
    for (int k = 0; k < K; k += 32) {
        if (k + 128 < K) __builtin_prefetch(a + k + 128, 0, 0);  // global_prefetch_b8
        v16bf af = load16(a + k);
#pragma unroll
        for (int t = 0; t < 4; ++t) {
            const TB* b = Bm + (int64_t)(t * 16 + r) * ldb + koff + k;
            if (k + 128 < K) __builtin_prefetch(b + 128, 0, 0);
            acc[t] = wmma_bf16(af, load16(b), acc[t]);
        }
    }
}
// C/D f32 16x16 layout: element e of lane l -> row = e + 8*(l>>4), col = l&15.

// ---------------- kernels ----------------

// vectorized fp32 -> bf16 (8 elems / thread); n8 = count of 8-elem chunks
__global__ void __launch_bounds__(256)
k_cvt_bf16(const float* __restrict__ x, bf16_t* __restrict__ y, int n8) {
    int i = blockIdx.x * blockDim.x + threadIdx.x;
    if (i < n8) {
        v8f f = *(const v8f*)(x + (int64_t)i * 8);
        *(v8bf*)(y + (int64_t)i * 8) = __builtin_convertvector(f, v8bf);
    }
}

// Y[NTOK, EMB] = X[NTOK, EMB] @ W[EMB, EMB]^T + bias   (fp32 out, W pre-bf16)
template <typename TA>
__global__ void __launch_bounds__(128)
k_linear(const TA* __restrict__ X, const bf16_t* __restrict__ W,
         const float* __restrict__ bias, float* __restrict__ Y) {
    const int wv = threadIdx.x >> 5;
    const int m0 = (blockIdx.y * 4 + wv) * 16;
    const int n0 = blockIdx.x * 64;
    v8f acc[4] = {};
    wave_gemm(X + (int64_t)m0 * EMB, EMB, W + (int64_t)n0 * EMB, EMB, EMB, acc);
    const int lane = threadIdx.x & 31;
    const int half = lane >> 4;
    const int col  = lane & 15;
#pragma unroll
    for (int t = 0; t < 4; ++t) {
        const int cc = n0 + t * 16 + col;
        const float bc = bias[cc];
#pragma unroll
        for (int e = 0; e < 8; ++e) {
            int rr = m0 + e + 8 * half;
            Y[(int64_t)rr * EMB + cc] = acc[t][e] + bc;
        }
    }
}

// row-wise L2 normalize [NTOK, EMB] fp32 -> bf16 (4 elems / thread)
__global__ void __launch_bounds__(256)
k_l2norm(const float* __restrict__ X, bf16_t* __restrict__ Y) {
    __shared__ float red[256];
    const int64_t row = blockIdx.x;
    const float* x = X + row * EMB;
    const int i4 = threadIdx.x * 4;                 // 256*4 == EMB
    v4f xv = *(const v4f*)(x + i4);
    red[threadIdx.x] = xv[0]*xv[0] + xv[1]*xv[1] + xv[2]*xv[2] + xv[3]*xv[3];
    __syncthreads();
    for (int off = 128; off; off >>= 1) {
        if (threadIdx.x < off) red[threadIdx.x] += red[threadIdx.x + off];
        __syncthreads();
    }
    const float inv = 1.0f / sqrtf(red[0]);
    *(v4bf*)(Y + row * EMB + i4) = __builtin_convertvector(xv * inv, v4bf);
}

// vT[b][h][d][kv] = bf16( vf[b][kv][h*HD + d] )
__global__ void __launch_bounds__(256)
k_vtranspose(const float* __restrict__ vf, bf16_t* __restrict__ vT) {
    int64_t o = (int64_t)blockIdx.x * blockDim.x + threadIdx.x;
    if (o >= (int64_t)BATCH * NH * HD * KVLEN) return;
    int kv = (int)(o % KVLEN);
    int d  = (int)((o / KVLEN) % HD);
    int h  = (int)((o / ((int64_t)KVLEN * HD)) % NH);
    int b  = (int)(o / ((int64_t)KVLEN * HD * NH));
    vT[o] = f2bf(vf[((int64_t)b * KVLEN + kv) * EMB + h * HD + d]);
}

// Fused per-(head, 16-query-row strip): scores -> LDS, softmax in LDS,
// write fp32 probs to d_out ONCE, then O = P @ V^T straight from LDS.
// One wave per block; 16 x KVLEN bf16 scores = 64 KB LDS.
__global__ void __launch_bounds__(32)
k_attn_fused(const bf16_t* __restrict__ qb, const bf16_t* __restrict__ kb,
             const bf16_t* __restrict__ vT, const float* __restrict__ lscale,
             const float* __restrict__ lbias, float* __restrict__ attn,
             bf16_t* __restrict__ Ob) {
    __shared__ bf16_t sP[16 * KVLEN];          // 64 KB
    const int z = blockIdx.y;                  // b*NH + h
    const int b = z / NH, h = z % NH;
    const int m0 = blockIdx.x * 16;            // query row strip
    const int lane = threadIdx.x;
    const int r = lane & 15;
    const int half = lane >> 4;
    const int koff = half << 3;

    const float scale = __expf(lscale[0]);
    const float sbias = lbias[0];

    // ---- stage 1: S[16, KVLEN] = (q strip) @ k^T * scale + bias -> LDS bf16
    const bf16_t* Aq = qb + ((int64_t)b * QLEN + m0) * EMB + h * HD;
    const bf16_t* Bk = kb + (int64_t)b * KVLEN * EMB + h * HD;
    const bf16_t* arow = Aq + (int64_t)r * EMB + koff;
    v16bf af0 = load16(arow);        // K = 0..31 of HD
    v16bf af1 = load16(arow + 32);   // K = 32..63
    for (int n0 = 0; n0 < KVLEN; n0 += 64) {
        v8f acc[4] = {};
#pragma unroll
        for (int t = 0; t < 4; ++t) {
            const bf16_t* brow = Bk + (int64_t)(n0 + t * 16 + r) * EMB + koff;
            acc[t] = wmma_bf16(af0, load16(brow), acc[t]);
            acc[t] = wmma_bf16(af1, load16(brow + 32), acc[t]);
        }
#pragma unroll
        for (int t = 0; t < 4; ++t) {
            v8bf sb = __builtin_convertvector(acc[t] * scale + sbias, v8bf);
#pragma unroll
            for (int e = 0; e < 8; ++e) {
                int rr = e + 8 * half;
                int cc = n0 + t * 16 + (lane & 15);
                sP[rr * KVLEN + cc] = sb[e];
            }
        }
    }
    __syncthreads();

    // ---- stage 2: softmax per row (f32 math), probs -> global f32 + LDS bf16
    float* attn_base = attn + ((int64_t)z * QLEN + m0) * KVLEN;
    for (int rr = 0; rr < 16; ++rr) {
        bf16_t* srow = sP + rr * KVLEN;
        v8bf* vrow = (v8bf*)srow;                 // KVLEN/8 = 256 chunks
        float m = -3.402823466e38f;
#pragma unroll
        for (int j = 0; j < 8; ++j) {
            v8f cf = __builtin_convertvector(vrow[lane + 32 * j], v8f);
#pragma unroll
            for (int t = 0; t < 8; ++t) m = fmaxf(m, cf[t]);
        }
        for (int off = 16; off; off >>= 1) m = fmaxf(m, __shfl_xor(m, off));
        float s = 0.f;
#pragma unroll
        for (int j = 0; j < 8; ++j) {
            v8f cf = __builtin_convertvector(vrow[lane + 32 * j], v8f);
#pragma unroll
            for (int t = 0; t < 8; ++t) { cf[t] = __expf(cf[t] - m); s += cf[t]; }
            vrow[lane + 32 * j] = __builtin_convertvector(cf, v8bf);  // unnormalized
        }
        for (int off = 16; off; off >>= 1) s += __shfl_xor(s, off);
        const float inv = 1.0f / s;
        float* grow = attn_base + (int64_t)rr * KVLEN;
#pragma unroll
        for (int j = 0; j < 8; ++j) {
            int ci = lane + 32 * j;
            v8f cf = __builtin_convertvector(vrow[ci], v8f) * inv;
            *(v8f*)(grow + ci * 8) = cf;          // fp32 attn output (once)
            vrow[ci] = __builtin_convertvector(cf, v8bf);  // normalized bf16
        }
    }
    __syncthreads();

    // ---- stage 3: O[16, HD] = P[16, KVLEN] @ V^T (A-frags from LDS)
    const bf16_t* Bv = vT + (int64_t)z * HD * KVLEN;
    v8f acc[4] = {};
    const bf16_t* prow = sP + r * KVLEN + koff;
    for (int k = 0; k < KVLEN; k += 32) {
        v16bf pf = load16(prow + k);              // ds_load_b128 x2
#pragma unroll
        for (int t = 0; t < 4; ++t) {
            const bf16_t* brow = Bv + (int64_t)(t * 16 + r) * KVLEN + koff + k;
            acc[t] = wmma_bf16(pf, load16(brow), acc[t]);
        }
    }
    const int col = lane & 15;
#pragma unroll
    for (int t = 0; t < 4; ++t) {
        v8bf ob = __builtin_convertvector(acc[t], v8bf);
#pragma unroll
        for (int e = 0; e < 8; ++e) {
            int rr = m0 + e + 8 * half;
            int cc = t * 16 + col;
            Ob[((int64_t)b * QLEN + rr) * EMB + h * HD + cc] = ob[e];
        }
    }
}

// ---------------- host launcher ----------------
extern "C" void kernel_launch(void* const* d_in, const int* in_sizes, int n_in,
                              void* d_out, int out_size, void* d_ws, size_t ws_size,
                              hipStream_t stream) {
    (void)in_sizes; (void)n_in; (void)out_size; (void)ws_size;
    const float* query  = (const float*)d_in[0];
    const float* key    = (const float*)d_in[1];
    const float* value  = (const float*)d_in[2];
    const float* lscale = (const float*)d_in[3];
    const float* lbias  = (const float*)d_in[4];
    const float* Wq = (const float*)d_in[5];
    const float* bq = (const float*)d_in[6];
    const float* Wk = (const float*)d_in[7];
    const float* bk = (const float*)d_in[8];
    const float* Wv = (const float*)d_in[9];
    const float* bv = (const float*)d_in[10];
    const float* Wo = (const float*)d_in[11];
    const float* bo = (const float*)d_in[12];

    float* out_p  = (float*)d_out;                          // [B, QL, EMB]
    float* attn_p = out_p + (size_t)BATCH * QLEN * EMB;     // [B, H, QL, KVL]

    const size_t TOKEMB = (size_t)NTOK * EMB;               // 4096*1024
    const size_t WELEM  = (size_t)EMB * EMB;                // 1024*1024
    float*  qf  = (float*)d_ws;
    float*  kf  = qf + TOKEMB;
    float*  vf  = kf + TOKEMB;
    bf16_t* qb  = (bf16_t*)(vf + TOKEMB);
    bf16_t* kb  = qb + TOKEMB;
    bf16_t* vT  = kb + TOKEMB;   // [B*NH, HD, KVLEN] == TOKEMB elems
    bf16_t* Ob  = vT + TOKEMB;   // [B, QL, EMB] bf16
    bf16_t* Wqb = Ob + TOKEMB;
    bf16_t* Wkb = Wqb + WELEM;
    bf16_t* Wvb = Wkb + WELEM;
    bf16_t* Wob = Wvb + WELEM;

    const dim3 blk128(128);

    // 0) pre-convert weights to bf16 (kills repeated cvt VALU in GEMM loops)
    {
        int n8 = (int)(WELEM / 8);
        dim3 g((unsigned)((n8 + 255) / 256));
        k_cvt_bf16<<<g, dim3(256), 0, stream>>>(Wq, Wqb, n8);
        k_cvt_bf16<<<g, dim3(256), 0, stream>>>(Wk, Wkb, n8);
        k_cvt_bf16<<<g, dim3(256), 0, stream>>>(Wv, Wvb, n8);
        k_cvt_bf16<<<g, dim3(256), 0, stream>>>(Wo, Wob, n8);
    }
    // 1) projections: q/k/v = X @ W^T + b  (fp32 out)
    {
        dim3 g(EMB / 64, NTOK / 64);
        k_linear<float><<<g, blk128, 0, stream>>>(query, Wqb, bq, qf);
        k_linear<float><<<g, blk128, 0, stream>>>(key,   Wkb, bk, kf);
        k_linear<float><<<g, blk128, 0, stream>>>(value, Wvb, bv, vf);
    }
    // 2) L2-normalize q, k rows over full EMB -> bf16
    k_l2norm<<<dim3(NTOK), dim3(256), 0, stream>>>(qf, qb);
    k_l2norm<<<dim3(NTOK), dim3(256), 0, stream>>>(kf, kb);
    // 3) per-head transpose of v -> [B,H,HD,KVLEN] bf16
    {
        int64_t n = (int64_t)BATCH * NH * HD * KVLEN;
        k_vtranspose<<<dim3((unsigned)((n + 255) / 256)), dim3(256), 0, stream>>>(vf, vT);
    }
    // 4) fused scores + softmax + attn@V: attn written once, probs kept in LDS
    k_attn_fused<<<dim3(QLEN / 16, BATCH * NH), dim3(32), 0, stream>>>(
        qb, kb, vT, lscale, lbias, attn_p, Ob);
    // 5) out = O @ Wo^T + bo (fp32)
    k_linear<bf16_t><<<dim3(EMB / 64, NTOK / 64), blk128, 0, stream>>>(
        Ob, Wob, bo, out_p);
}
